// AirGraph_23708219474018
// MI455X (gfx1250) — matrix operand order repaired and checked by reading, not compiled, not run on hardware
//
#include <hip/hip_runtime.h>
#include <hip/hip_bf16.h>

// ---------------------------------------------------------------------------
// Types for WMMA fragments (CDNA5 gfx1250, wave32)
// ---------------------------------------------------------------------------
typedef __attribute__((ext_vector_type(16))) __bf16       bf16x16;
typedef __attribute__((ext_vector_type(8)))  float        f32x8;
typedef __attribute__((ext_vector_type(4)))  unsigned int u32x4;
typedef __attribute__((ext_vector_type(4)))  float        f32x4;

union BFrag { u32x4 q[2]; bf16x16 v; };

__device__ __forceinline__ unsigned short f2bf(float f) {
    unsigned b = __float_as_uint(f);
    unsigned r = b + 0x7FFFu + ((b >> 16) & 1u);   // round to nearest even
    return (unsigned short)(r >> 16);
}

// monotone float<->uint encoding so atomicMax(u32) == float max
__device__ __forceinline__ unsigned encf(float f) {
    unsigned b = __float_as_uint(f);
    return (b & 0x80000000u) ? ~b : (b | 0x80000000u);
}
__device__ __forceinline__ float decf(unsigned u) {
    return (u & 0x80000000u) ? __uint_as_float(u & 0x7FFFFFFFu)
                             : __uint_as_float(~u);
}

__device__ __forceinline__ float wred(float v) {
    #pragma unroll
    for (int o = 16; o; o >>= 1) v += __shfl_xor(v, o, 32);
    return v;
}

// ---------------------------------------------------------------------------
// Pack fp32 weight matrix (K x 128, row-major) into bf16 WMMA B-fragment
// order.  For tile (kt, nt): lane l holds column (l&15), K = (l>>4)*16 + j,
// stored contiguously so the GEMM does two 16B loads per fragment.
// ---------------------------------------------------------------------------
__global__ void pack_b(const float* __restrict__ W, unsigned short* __restrict__ dst,
                       int K, int KP, int srcLd, int nTilesTotal, int colTileOff) {
    int idx   = blockIdx.x * blockDim.x + threadIdx.x;
    int total = (KP >> 5) * 8 * 512;   // 8 local 16-col tiles per 128-col source
    if (idx >= total) return;
    int j   = idx & 15;
    int l   = (idx >> 4) & 31;
    int ntL = (idx >> 9) & 7;
    int kt  = idx >> 12;
    int k   = (kt << 5) + ((l >> 4) * 16 + j);
    int n   = ntL * 16 + (l & 15);
    float v = (k < K) ? W[(size_t)k * srcLd + n] : 0.0f;
    size_t o = ((size_t)kt * nTilesTotal + colTileOff + ntL) * 512 + (size_t)l * 16 + j;
    dst[o] = f2bf(v);
}

// fp32 (M x K) -> bf16 (Mpad x KP) with zero padding
__global__ void conv_bf16(const float* __restrict__ A, unsigned short* __restrict__ out,
                          int M, int K, int KP, int total) {
    int idx = blockIdx.x * blockDim.x + threadIdx.x;
    if (idx >= total) return;
    int r = idx / KP, k = idx - r * KP;
    float v = (r < M && k < K) ? A[(size_t)r * K + k] : 0.0f;
    out[idx] = f2bf(v);
}

__global__ void cat_bias(float* __restrict__ dst, const float* a, const float* b,
                         const float* c, int n1, int n2, int n3) {
    int i = blockIdx.x * blockDim.x + threadIdx.x;
    if (i < n1) dst[i] = a[i];
    else if (i < n1 + n2) dst[i] = b[i - n1];
    else if (i < n1 + n2 + n3) dst[i] = c[i - n1 - n2];
}

__global__ void fill_u32(unsigned* __restrict__ p, unsigned v, int n) {
    int i = blockIdx.x * blockDim.x + threadIdx.x;
    if (i < n) p[i] = v;
}

// ---------------------------------------------------------------------------
// bf16 WMMA GEMM:  C[M x Nout] = Abf[Mpad x KP] @ Bpacked + bias
// A pre-converted/padded bf16; B packed in fragment order.
// Block: 128 threads = 4 waves; each wave computes a 16x64 tile.
// Software-pipelined: tile k+1 loads are issued before tile k's wmmas so the
// VMEM pipe overlaps the matrix pipe (partial s_wait_loadcnt, not 0).
// ---------------------------------------------------------------------------
__device__ __forceinline__ void load_tile(const unsigned short* __restrict__ aRow,
                                          const unsigned short* __restrict__ bBase,
                                          size_t bStride, int kt,
                                          BFrag& a, BFrag (&b)[4]) {
    const unsigned short* ap = aRow + ((size_t)kt << 5);
    a.q[0] = *reinterpret_cast<const u32x4*>(ap);
    a.q[1] = *reinterpret_cast<const u32x4*>(ap + 8);
    const unsigned short* bp = bBase + (size_t)kt * bStride;
    #pragma unroll
    for (int nf = 0; nf < 4; ++nf) {
        b[nf].q[0] = *reinterpret_cast<const u32x4*>(bp + nf * 512);
        b[nf].q[1] = *reinterpret_cast<const u32x4*>(bp + nf * 512 + 8);
    }
}

__device__ __forceinline__ void mma4(f32x8 (&acc)[4], const BFrag& a, const BFrag (&b)[4]) {
    #pragma unroll
    for (int nf = 0; nf < 4; ++nf)
        acc[nf] = __builtin_amdgcn_wmma_f32_16x16x32_bf16(
            false, a.v, false, b[nf].v, (short)0, acc[nf], false, false);
}

__global__ void __launch_bounds__(128)
gemm_bf16(const unsigned short* __restrict__ Abf, int lda,
          const unsigned short* __restrict__ Bp,
          const float* __restrict__ bias, float* __restrict__ C,
          int M, int KP, int Nout, int ldc) {
    const int wave = threadIdx.x >> 5, lane = threadIdx.x & 31;
    const int mw   = blockIdx.x * 64 + wave * 16;
    const int nT16 = Nout >> 4;
    const int nBlk = blockIdx.y * 4;
    const int kh   = lane >> 4;
    const int rowA = mw + (lane & 15);
    const unsigned short* aRow  = Abf + (size_t)rowA * lda + kh * 16;
    const unsigned short* bBase = Bp + (size_t)nBlk * 512 + (size_t)lane * 16;
    const size_t bStride = (size_t)nT16 * 512;
    const int KT = KP >> 5;

    f32x8 acc[4] = {};
    BFrag aX, bX[4], aY, bY[4];

    load_tile(aRow, bBase, bStride, 0, aX, bX);
    int kt = 0;
    for (; kt + 2 <= KT; kt += 2) {
        load_tile(aRow, bBase, bStride, kt + 1, aY, bY);   // prefetch
        mma4(acc, aX, bX);
        if (kt + 2 < KT) load_tile(aRow, bBase, bStride, kt + 2, aX, bX);
        mma4(acc, aY, bY);
    }
    if (kt < KT) mma4(acc, aX, bX);                        // odd-KT tail

    const int colBase = blockIdx.y * 64;
    const int rBase   = mw + kh * 8;
    #pragma unroll
    for (int nf = 0; nf < 4; ++nf) {
        int col  = colBase + nf * 16 + (lane & 15);
        float bv = bias ? bias[col] : 0.0f;
        #pragma unroll
        for (int r = 0; r < 8; ++r) {
            int row = rBase + r;
            if (row < M) C[(size_t)row * ldc + col] = acc[nf][r] + bv;
        }
    }
}

// ---------------------------------------------------------------------------
// Edge pass 1: per-edge GATv2 score (ee computed on the fly from LDS-staged
// We) + segment max.  One wave32 per edge, lane -> channel (4 heads x 32 ch).
// ---------------------------------------------------------------------------
__global__ void __launch_bounds__(256)
edge_score(const float* __restrict__ nodeT, int st, int xrOff,
           const float* __restrict__ ea, const float* __restrict__ We,
           const float* __restrict__ att,
           const int* __restrict__ src, const int* __restrict__ dst, int E,
           float* __restrict__ score, unsigned* __restrict__ segmax) {
    __shared__ float sWe[16 * 128];
    __shared__ float sAtt[128];
    for (int i = threadIdx.x; i < 16 * 128; i += 256) sWe[i] = We[i];
    if (threadIdx.x < 128) sAtt[threadIdx.x] = att[threadIdx.x];
    __syncthreads();

    int wave = threadIdx.x >> 5, lane = threadIdx.x & 31;
    int e = blockIdx.x * 8 + wave;
    if (e >= E) return;
    int s = src[e], d = dst[e];
    const float* xl = nodeT + (size_t)s * st;
    const float* xr = nodeT + (size_t)d * st + xrOff;

    const f32x4* eap = reinterpret_cast<const f32x4*>(ea + (size_t)e * 16);
    f32x4 e0 = eap[0], e1 = eap[1], e2 = eap[2], e3 = eap[3];
    float eav[16];
    #pragma unroll
    for (int k = 0; k < 4; ++k) {
        eav[k] = e0[k]; eav[4 + k] = e1[k]; eav[8 + k] = e2[k]; eav[12 + k] = e3[k];
    }

    float p[4];
    #pragma unroll
    for (int i = 0; i < 4; ++i) {
        int ch = i * 32 + lane;
        float ee = 0.f;
        #pragma unroll
        for (int k = 0; k < 16; ++k) ee = fmaf(eav[k], sWe[k * 128 + ch], ee);
        float m  = xl[ch] + xr[ch] + ee;
        float gl = m > 0.f ? m : 0.2f * m;       // leaky_relu(0.2)
        p[i] = gl * sAtt[ch];                     // att[h*32+c]
    }
    #pragma unroll
    for (int i = 0; i < 4; ++i) p[i] = wred(p[i]);
    if (lane == 0) {
        #pragma unroll
        for (int i = 0; i < 4; ++i) {
            score[(size_t)e * 4 + i] = p[i];
            atomicMax(&segmax[(size_t)d * 4 + i], encf(p[i]));
        }
    }
}

// Edge pass 2: ex = exp(score - max), segment sum
__global__ void edge_softmax(float* __restrict__ alpha, const int* __restrict__ dst,
                             const unsigned* __restrict__ segmax,
                             float* __restrict__ segsum, int E) {
    int idx = blockIdx.x * blockDim.x + threadIdx.x;
    if (idx >= E * 4) return;
    int e = idx >> 2, h = idx & 3;
    int d = dst[e];
    float m  = decf(segmax[(size_t)d * 4 + h]);
    float ex = __expf(alpha[idx] - m);
    alpha[idx] = ex;
    atomicAdd(&segsum[(size_t)d * 4 + h], ex);
}

// Edge pass 3: scatter alpha-weighted xl[src] into agg[dst]
__global__ void __launch_bounds__(256)
edge_message(const float* __restrict__ nodeT, int st,
             const float* __restrict__ alpha, const float* __restrict__ segsum,
             const int* __restrict__ src, const int* __restrict__ dst, int E,
             float* __restrict__ agg) {
    int wave = threadIdx.x >> 5, lane = threadIdx.x & 31;
    int e = blockIdx.x * 8 + wave;
    if (e >= E) return;
    int s = src[e], d = dst[e];
    #pragma unroll
    for (int i = 0; i < 4; ++i) {
        float w  = alpha[(size_t)e * 4 + i] / (segsum[(size_t)d * 4 + i] + 1e-16f);
        int   ch = i * 32 + lane;
        atomicAdd(&agg[(size_t)d * 128 + ch], nodeT[(size_t)s * st + ch] * w);
    }
}

// ---------------------------------------------------------------------------
// Node epilogue: h = LN(elu(agg + bias) + residual) * g + b ; wave per node.
// Emits f32 (optional) and bf16 (optional, feeds the next WMMA GEMM).
// ---------------------------------------------------------------------------
__global__ void __launch_bounds__(256)
gat_epilogue(const float* __restrict__ agg, const float* __restrict__ bias,
             const float* __restrict__ res, int resLd, int resOff,
             const float* __restrict__ g, const float* __restrict__ b,
             float* __restrict__ out, unsigned short* __restrict__ outbf, int Nn) {
    int wave = threadIdx.x >> 5, lane = threadIdx.x & 31;
    int n = blockIdx.x * 8 + wave;
    if (n >= Nn) return;
    float v[4];
    #pragma unroll
    for (int i = 0; i < 4; ++i) {
        int ch = i * 32 + lane;
        float x = agg[(size_t)n * 128 + ch] + bias[ch];
        x = x > 0.f ? x : (expf(x) - 1.f);       // elu
        v[i] = x + res[(size_t)n * resLd + resOff + ch];
    }
    float mu = wred(v[0] + v[1] + v[2] + v[3]) * (1.f / 128.f);
    float q = 0.f;
    #pragma unroll
    for (int i = 0; i < 4; ++i) { float dd = v[i] - mu; q += dd * dd; }
    float inv = rsqrtf(wred(q) * (1.f / 128.f) + 1e-5f);
    #pragma unroll
    for (int i = 0; i < 4; ++i) {
        int ch = i * 32 + lane;
        float y = (v[i] - mu) * inv * g[ch] + b[ch];
        if (out)   out[(size_t)n * 128 + ch] = y;
        if (outbf) outbf[(size_t)n * 128 + ch] = f2bf(y);
    }
}

__global__ void __launch_bounds__(256)
ln_final(const float* __restrict__ in, const float* __restrict__ g,
         const float* __restrict__ b, float* __restrict__ out, int Nn) {
    int wave = threadIdx.x >> 5, lane = threadIdx.x & 31;
    int n = blockIdx.x * 8 + wave;
    if (n >= Nn) return;
    float v[4];
    #pragma unroll
    for (int i = 0; i < 4; ++i) v[i] = in[(size_t)n * 128 + i * 32 + lane];
    float mu = wred(v[0] + v[1] + v[2] + v[3]) * (1.f / 128.f);
    float q = 0.f;
    #pragma unroll
    for (int i = 0; i < 4; ++i) { float dd = v[i] - mu; q += dd * dd; }
    float inv = rsqrtf(wred(q) * (1.f / 128.f) + 1e-5f);
    #pragma unroll
    for (int i = 0; i < 4; ++i) {
        int ch = i * 32 + lane;
        out[(size_t)n * 128 + ch] = (v[i] - mu) * inv * g[ch] + b[ch];
    }
}

// ---------------------------------------------------------------------------
// Host-side orchestration
// ---------------------------------------------------------------------------
extern "C" void kernel_launch(void* const* d_in, const int* in_sizes, int n_in,
                              void* d_out, int out_size, void* d_ws, size_t ws_size,
                              hipStream_t stream) {
    (void)n_in; (void)out_size; (void)ws_size;
    const float* x     = (const float*)d_in[0];
    const float* ea    = (const float*)d_in[1];
    const float* Wl1   = (const float*)d_in[2];
    const float* bl1   = (const float*)d_in[3];
    const float* Wr1   = (const float*)d_in[4];
    const float* br1   = (const float*)d_in[5];
    const float* We1   = (const float*)d_in[6];
    const float* att1  = (const float*)d_in[7];
    const float* bias1 = (const float*)d_in[8];
    const float* Wl2   = (const float*)d_in[9];
    const float* bl2   = (const float*)d_in[10];
    const float* Wr2   = (const float*)d_in[11];
    const float* br2   = (const float*)d_in[12];
    const float* We2   = (const float*)d_in[13];
    const float* att2  = (const float*)d_in[14];
    const float* bias2 = (const float*)d_in[15];
    const float* Wres  = (const float*)d_in[16];
    const float* bres  = (const float*)d_in[17];
    const float* Wout  = (const float*)d_in[18];
    const float* bout  = (const float*)d_in[19];
    const float* g1    = (const float*)d_in[20];
    const float* bn1   = (const float*)d_in[21];
    const float* g2    = (const float*)d_in[22];
    const float* bn2   = (const float*)d_in[23];
    const float* go    = (const float*)d_in[24];
    const float* bo    = (const float*)d_in[25];
    const int*   ei    = (const int*)d_in[26];

    const int N = in_sizes[0] / 264;
    const int E = in_sizes[26] / 2;
    const int Mpad = (N + 63) & ~63;
    const int* src = ei;
    const int* dst = ei + E;

    // workspace carve-out
    char* ws = (char*)d_ws;
    size_t off = 0;
    auto carve = [&](size_t bytes) -> char* {
        char* p = ws + off;
        off += (bytes + 255) & ~(size_t)255;
        return p;
    };
    unsigned short* W1p  = (unsigned short*)carve((size_t)288 * 384 * 2);
    float*          b1c  = (float*)carve(384 * 4);
    unsigned short* W2p  = (unsigned short*)carve((size_t)128 * 256 * 2);
    float*          b2c  = (float*)carve(256 * 4);
    unsigned short* Wop  = (unsigned short*)carve((size_t)128 * 128 * 2);
    unsigned short* xbf  = (unsigned short*)carve((size_t)Mpad * 288 * 2);
    unsigned short* h1bf = (unsigned short*)carve((size_t)Mpad * 128 * 2);
    unsigned short* h2bf = (unsigned short*)carve((size_t)Mpad * 128 * 2);
    float* nodeT1 = (float*)carve((size_t)N * 384 * 4);   // [xl1|xr1|res]
    float* agg    = (float*)carve((size_t)N * 128 * 4);
    float* h1     = (float*)carve((size_t)N * 128 * 4);
    float* nodeT2 = (float*)carve((size_t)N * 256 * 4);   // [xl2|xr2]
    float* alpha  = (float*)carve((size_t)E * 4 * 4);
    unsigned* segmax = (unsigned*)carve((size_t)N * 4 * 4);
    float*    segsum = (float*)carve((size_t)N * 4 * 4);
    float* ctx = nodeT2;   // reuse (free after layer-2 score/message)

    const unsigned ENC_NEG_INF = 0x007FFFFFu;   // encf(-inf)

    // ---- pack weights (bf16 WMMA fragment order), concat biases, convert x ----
    {
        int tot1 = (288 >> 5) * 8 * 512;
        pack_b<<<(tot1 + 255) / 256, 256, 0, stream>>>(Wl1,  W1p, 264, 288, 128, 24, 0);
        pack_b<<<(tot1 + 255) / 256, 256, 0, stream>>>(Wr1,  W1p, 264, 288, 128, 24, 8);
        pack_b<<<(tot1 + 255) / 256, 256, 0, stream>>>(Wres, W1p, 264, 288, 128, 24, 16);
        int tot2 = (128 >> 5) * 8 * 512;
        pack_b<<<(tot2 + 255) / 256, 256, 0, stream>>>(Wl2,  W2p, 128, 128, 128, 16, 0);
        pack_b<<<(tot2 + 255) / 256, 256, 0, stream>>>(Wr2,  W2p, 128, 128, 128, 16, 8);
        pack_b<<<(tot2 + 255) / 256, 256, 0, stream>>>(Wout, Wop, 128, 128, 128, 8, 0);
        cat_bias<<<2, 256, 0, stream>>>(b1c, bl1, br1, bres, 128, 128, 128);
        cat_bias<<<1, 256, 0, stream>>>(b2c, bl2, br2, nullptr, 128, 128, 0);

        int totX = Mpad * 288;
        conv_bf16<<<(totX + 255) / 256, 256, 0, stream>>>(x, xbf, N, 264, 288, totX);

        // zero the padded tail rows of h1bf / h2bf (they are read by WMMA loads)
        int tail = (Mpad - N) * 128;           // #u16 elements
        if (tail > 0) {
            fill_u32<<<(tail / 2 + 255) / 256, 256, 0, stream>>>(
                (unsigned*)(h1bf + (size_t)N * 128), 0u, tail / 2);
            fill_u32<<<(tail / 2 + 255) / 256, 256, 0, stream>>>(
                (unsigned*)(h2bf + (size_t)N * 128), 0u, tail / 2);
        }
    }

    // ---- layer 1 ----
    fill_u32<<<((size_t)N * 128 + 255) / 256, 256, 0, stream>>>((unsigned*)agg, 0u, N * 128);
    fill_u32<<<(N * 4 + 255) / 256, 256, 0, stream>>>((unsigned*)segsum, 0u, N * 4);
    fill_u32<<<(N * 4 + 255) / 256, 256, 0, stream>>>(segmax, ENC_NEG_INF, N * 4);

    dim3 gg1(Mpad / 64, 6);
    gemm_bf16<<<gg1, 128, 0, stream>>>(xbf, 288, W1p, b1c, nodeT1, N, 288, 384, 384);

    edge_score  <<<(E + 7) / 8, 256, 0, stream>>>(nodeT1, 384, 128, ea, We1, att1,
                                                  src, dst, E, alpha, segmax);
    edge_softmax<<<(E * 4 + 255) / 256, 256, 0, stream>>>(alpha, dst, segmax, segsum, E);
    edge_message<<<(E + 7) / 8, 256, 0, stream>>>(nodeT1, 384, alpha, segsum,
                                                  src, dst, E, agg);
    gat_epilogue<<<(N + 7) / 8, 256, 0, stream>>>(agg, bias1, nodeT1, 384, 256,
                                                  g1, bn1, h1, h1bf, N);

    // ---- layer 2 ----
    fill_u32<<<((size_t)N * 128 + 255) / 256, 256, 0, stream>>>((unsigned*)agg, 0u, N * 128);
    fill_u32<<<(N * 4 + 255) / 256, 256, 0, stream>>>((unsigned*)segsum, 0u, N * 4);
    fill_u32<<<(N * 4 + 255) / 256, 256, 0, stream>>>(segmax, ENC_NEG_INF, N * 4);

    dim3 gg2(Mpad / 64, 4);
    gemm_bf16<<<gg2, 128, 0, stream>>>(h1bf, 128, W2p, b2c, nodeT2, N, 128, 256, 256);

    edge_score  <<<(E + 7) / 8, 256, 0, stream>>>(nodeT2, 256, 128, ea, We2, att2,
                                                  src, dst, E, alpha, segmax);
    edge_softmax<<<(E * 4 + 255) / 256, 256, 0, stream>>>(alpha, dst, segmax, segsum, E);
    edge_message<<<(E + 7) / 8, 256, 0, stream>>>(nodeT2, 256, alpha, segsum,
                                                  src, dst, E, agg);
    gat_epilogue<<<(N + 7) / 8, 256, 0, stream>>>(agg, bias2, h1, 128, 0,
                                                  g2, bn2, nullptr, h2bf, N);

    // ---- output projection + final LN ----
    dim3 gg3(Mpad / 64, 2);
    gemm_bf16<<<gg3, 128, 0, stream>>>(h2bf, 128, Wop, bout, ctx, N, 128, 128, 128);
    ln_final<<<(N + 7) / 8, 256, 0, stream>>>(ctx, go, bo, (float*)d_out, N);
}
